// CrossAttention_29583734735210
// MI455X (gfx1250) — compile-verified, matrix operand order checked by reading, not run on hardware
//
#include <hip/hip_runtime.h>

typedef __attribute__((ext_vector_type(16))) __bf16       v16bf;
typedef __attribute__((ext_vector_type(8)))  float        v8f;
typedef __attribute__((ext_vector_type(4)))  unsigned int v4u;

union FragU { v4u u[2]; v16bf b; };

__device__ __forceinline__ unsigned short f2bf(float x) {
  unsigned u = __builtin_bit_cast(unsigned, x);
  u += 0x7FFFu + ((u >> 16) & 1u);          // round-to-nearest-even
  return (unsigned short)(u >> 16);
}
__device__ __forceinline__ float bf2f(unsigned short h) {
  unsigned u = ((unsigned)h) << 16;
  return __builtin_bit_cast(float, u);
}
__device__ __forceinline__ v8f wmma_bf16(v16bf a, v16bf b, v8f c) {
  return __builtin_amdgcn_wmma_f32_16x16x32_bf16(false, a, false, b, (short)0, c,
                                                 false, false);
}

// ---------------------------------------------------------------------------
// Weight transpose + fp32 -> bf16.  wt[n*K + k] = bf16(w[k*N + n])
// ---------------------------------------------------------------------------
__global__ __launch_bounds__(256) void k_wT(const float* __restrict__ w,
                                            unsigned short* __restrict__ wt,
                                            int K, int N) {
  int idx = blockIdx.x * 256 + threadIdx.x;
  if (idx >= K * N) return;
  int n = idx / K, kk = idx % K;
  wt[(size_t)n * K + kk] = f2bf(w[(size_t)kk * N + n]);
}

// ---------------------------------------------------------------------------
// Pack byte mask (B,NA,NB) -> bit mask (B,NA,NB/32) uint32.
// ---------------------------------------------------------------------------
__global__ __launch_bounds__(256) void k_packmask(
    const unsigned char* __restrict__ m, unsigned int* __restrict__ pm,
    int total_words) {
  int idx = blockIdx.x * 256 + threadIdx.x;
  if (idx >= total_words) return;
  const unsigned char* src = m + (size_t)idx * 32;
  unsigned int bits = 0;
#pragma unroll
  for (int e = 0; e < 32; e++) bits |= (src[e] ? 1u : 0u) << e;
  pm[idx] = bits;
}

// ---------------------------------------------------------------------------
// LayerNorm over D=256, fp32 in -> bf16 out, one row per wave (wave32).
// ---------------------------------------------------------------------------
__global__ __launch_bounds__(128) void k_ln_feat(const float* __restrict__ x,
                                                 const float* __restrict__ w,
                                                 const float* __restrict__ bvec,
                                                 unsigned short* __restrict__ y,
                                                 int rows) {
  int lane = threadIdx.x & 31, wv = threadIdx.x >> 5;
  int row = blockIdx.x * 4 + wv;
  if (row >= rows) return;
  const float* xr = x + (size_t)row * 256;
  float vals[8];
  float s = 0.f;
#pragma unroll
  for (int e = 0; e < 8; e++) { vals[e] = xr[lane + 32 * e]; s += vals[e]; }
#pragma unroll
  for (int o_ = 16; o_ > 0; o_ >>= 1) s += __shfl_xor(s, o_, 32);
  float mean = s * (1.0f / 256.0f);
  float vs = 0.f;
#pragma unroll
  for (int e = 0; e < 8; e++) { float d = vals[e] - mean; vs += d * d; }
#pragma unroll
  for (int o_ = 16; o_ > 0; o_ >>= 1) vs += __shfl_xor(vs, o_, 32);
  float rstd = rsqrtf(vs * (1.0f / 256.0f) + 1e-5f);
#pragma unroll
  for (int e = 0; e < 8; e++) {
    int c = lane + 32 * e;
    y[(size_t)row * 256 + c] = f2bf((vals[e] - mean) * rstd * w[c] + bvec[c]);
  }
}

// ---------------------------------------------------------------------------
// LayerNorm over D=512 of pre-activation q/k, scatter to (B,H,N,64) bf16.
// ---------------------------------------------------------------------------
__global__ __launch_bounds__(128) void k_ln_qk(const float* __restrict__ pre,
                                               const float* __restrict__ w,
                                               const float* __restrict__ bvec,
                                               unsigned short* __restrict__ out,
                                               int rows) {
  int lane = threadIdx.x & 31, wv = threadIdx.x >> 5;
  int row = blockIdx.x * 4 + wv;
  if (row >= rows) return;
  const float* xr = pre + (size_t)row * 512;
  float vals[16];
  float s = 0.f;
#pragma unroll
  for (int e = 0; e < 16; e++) { vals[e] = xr[lane + 32 * e]; s += vals[e]; }
#pragma unroll
  for (int o_ = 16; o_ > 0; o_ >>= 1) s += __shfl_xor(s, o_, 32);
  float mean = s * (1.0f / 512.0f);
  float vs = 0.f;
#pragma unroll
  for (int e = 0; e < 16; e++) { float d = vals[e] - mean; vs += d * d; }
#pragma unroll
  for (int o_ = 16; o_ > 0; o_ >>= 1) vs += __shfl_xor(vs, o_, 32);
  float rstd = rsqrtf(vs * (1.0f / 512.0f) + 1e-5f);
  int b = row >> 11, i = row & 2047;
#pragma unroll
  for (int e = 0; e < 16; e++) {
    int c = lane + 32 * e;
    int h = c >> 6, d = c & 63;
    out[((size_t)(b * 8 + h) * 2048 + i) * 64 + d] =
        f2bf((vals[e] - mean) * rstd * w[c] + bvec[c]);
  }
}

// ---------------------------------------------------------------------------
// bf16 GEMM, 32x32 tile per wave (2x2 sub-tiles), f32 accumulate.
// K-loop unrolled by 2 with two *named* buffer sets (compile-time selection,
// no runtime-indexed arrays -> everything stays in VGPRs).
// mode 0: out f32 (M,N) = acc + bias
// mode 1: V path: scatter bf16 into vT (B,H,64,NB)
// mode 2: G path: scatter bf16 sigmoid(acc+bias) into (B,H,NA,64)
// ---------------------------------------------------------------------------
#define LOADK(kc, A0, A1, B0, B1)                                            \
  do {                                                                       \
    int o_ld = (kc) * 32;                                                    \
    A0.u[0] = *(const v4u*)(arow0 + o_ld);                                   \
    A0.u[1] = *(const v4u*)(arow0 + o_ld + 16);                              \
    A1.u[0] = *(const v4u*)(arow1 + o_ld);                                   \
    A1.u[1] = *(const v4u*)(arow1 + o_ld + 16);                              \
    B0.u[0] = *(const v4u*)(brow0 + o_ld);                                   \
    B0.u[1] = *(const v4u*)(brow0 + o_ld + 16);                              \
    B1.u[0] = *(const v4u*)(brow1 + o_ld);                                   \
    B1.u[1] = *(const v4u*)(brow1 + o_ld + 16);                              \
  } while (0)

__global__ __launch_bounds__(128) void k_gemm32(
    const unsigned short* __restrict__ A, const unsigned short* __restrict__ Bt,
    const float* __restrict__ bias, void* __restrict__ outp, int M, int N,
    int K, int mode) {
  int lane = threadIdx.x & 31;
  int wid = (int)((blockIdx.x * blockDim.x + threadIdx.x) >> 5);
  int nt = N >> 5;
  int tj = wid % nt, ti = wid / nt;
  if (ti >= (M >> 5)) return;
  int i0 = ti * 32, j0 = tj * 32;
  int hl = lane >> 4, l16 = lane & 15;

  const unsigned short* arow0 = A + (size_t)(i0 + l16) * K + 8 * hl;
  const unsigned short* arow1 = arow0 + (size_t)16 * K;
  const unsigned short* brow0 = Bt + (size_t)(j0 + l16) * K + 8 * hl;
  const unsigned short* brow1 = brow0 + (size_t)16 * K;

  v8f acc00, acc01, acc10, acc11;
#pragma unroll
  for (int e = 0; e < 8; e++) {
    acc00[e] = 0.f; acc01[e] = 0.f; acc10[e] = 0.f; acc11[e] = 0.f;
  }

  FragU pa0, pa1, pb0, pb1;  // ping buffers
  FragU qa0, qa1, qb0, qb1;  // pong buffers
  int ksteps = K >> 5;       // always even (K = 256 or 512)
  LOADK(0, pa0, pa1, pb0, pb1);
  for (int kc = 0; kc < ksteps; kc += 2) {
    LOADK(kc + 1, qa0, qa1, qb0, qb1);
    __builtin_prefetch(arow0 + (kc + 2) * 32, 0, 3);
    __builtin_prefetch(brow0 + (kc + 2) * 32, 0, 3);
    acc00 = wmma_bf16(pa0.b, pb0.b, acc00);
    acc01 = wmma_bf16(pa0.b, pb1.b, acc01);
    acc10 = wmma_bf16(pa1.b, pb0.b, acc10);
    acc11 = wmma_bf16(pa1.b, pb1.b, acc11);
    if (kc + 2 < ksteps) LOADK(kc + 2, pa0, pa1, pb0, pb1);
    acc00 = wmma_bf16(qa0.b, qb0.b, acc00);
    acc01 = wmma_bf16(qa0.b, qb1.b, acc01);
    acc10 = wmma_bf16(qa1.b, qb0.b, acc10);
    acc11 = wmma_bf16(qa1.b, qb1.b, acc11);
  }

  auto emit = [&](v8f accv, int ii, int jj) {
    int col = j0 + 16 * jj + l16;
    float bb = bias ? bias[col] : 0.f;
#pragma unroll
    for (int v = 0; v < 8; v++) {
      int row = i0 + 16 * ii + 8 * hl + v;
      float val = accv[v] + bb;
      if (mode == 0) {
        ((float*)outp)[(size_t)row * N + col] = val;
      } else if (mode == 1) {
        int b_ = row >> 11, j = row & 2047, h = col >> 6, d = col & 63;
        ((unsigned short*)outp)[((size_t)(b_ * 8 + h) * 64 + d) * 2048 + j] =
            f2bf(val);
      } else {
        int b_ = row >> 11, i = row & 2047, h = col >> 6, d = col & 63;
        float sgm = 1.0f / (1.0f + __expf(-val));
        ((unsigned short*)outp)[((size_t)(b_ * 8 + h) * 2048 + i) * 64 + d] =
            f2bf(sgm);
      }
    }
  };
  emit(acc00, 0, 0);
  emit(acc01, 0, 1);
  emit(acc10, 1, 0);
  emit(acc11, 1, 1);
}

// ---------------------------------------------------------------------------
// Flash attention. One wave: 16 Q rows of one (b,h); NB in steps of 32 with
// online softmax. q/k: (B,H,N,64) bf16; vt: (B,H,64,NB) bf16; pm: packed mask
// bits (B,NA,NB/32); sg: sigmoid gate (B,H,NA,64) bf16; o: (B*NA,512) bf16.
// ---------------------------------------------------------------------------
__global__ __launch_bounds__(128) void k_flash(
    const unsigned short* __restrict__ q, const unsigned short* __restrict__ k,
    const unsigned short* __restrict__ vt,
    const unsigned short* __restrict__ sg,
    const unsigned int* __restrict__ pm, unsigned short* __restrict__ o) {
  __shared__ __align__(16) unsigned short plds[4][16 * 32];
  int lane = threadIdx.x & 31, wv = threadIdx.x >> 5;
  int wid = (int)blockIdx.x * 4 + wv;
  int it = wid & 127;  // NA/16 = 128 tiles
  int bh = wid >> 7;   // 0..15
  int h = bh & 7, b = bh >> 3;
  int i0 = it * 16;
  int hl = lane >> 4, l16 = lane & 15;

  const unsigned short* qrow =
      q + ((size_t)(b * 8 + h) * 2048 + i0 + l16) * 64 + 8 * hl;
  FragU qf0, qf1;
  qf0.u[0] = *(const v4u*)(qrow);
  qf0.u[1] = *(const v4u*)(qrow + 16);
  qf1.u[0] = *(const v4u*)(qrow + 32);
  qf1.u[1] = *(const v4u*)(qrow + 48);

  v8f acc0, acc1, acc2, acc3;
#pragma unroll
  for (int e = 0; e < 8; e++) {
    acc0[e] = 0.f; acc1[e] = 0.f; acc2[e] = 0.f; acc3[e] = 0.f;
  }
  float mrow[8], lrow[8];
#pragma unroll
  for (int v = 0; v < 8; v++) { mrow[v] = -1e30f; lrow[v] = 0.f; }

  const unsigned short* kb = k + (size_t)(b * 8 + h) * 2048 * 64;
  const unsigned short* vb = vt + (size_t)(b * 8 + h) * 64 * 2048;
  const unsigned int* pmrow = pm + ((size_t)b * 2048 + i0) * 64;
  const float scale = 0.125f;  // 64^-0.5

  for (int j0 = 0; j0 < 2048; j0 += 32) {
    // ---- S = Q @ K^T for 16x32 strip (two 16x16 C-frags) ----
    v8f s[2];
#pragma unroll
    for (int t = 0; t < 2; t++) {
      const unsigned short* krow =
          kb + (size_t)(j0 + 16 * t + l16) * 64 + 8 * hl;
      __builtin_prefetch(krow + 32 * 64, 0, 3);  // next j-block of K
      FragU k0, k1;
      k0.u[0] = *(const v4u*)(krow);
      k0.u[1] = *(const v4u*)(krow + 16);
      k1.u[0] = *(const v4u*)(krow + 32);
      k1.u[1] = *(const v4u*)(krow + 48);
      v8f z;
#pragma unroll
      for (int e = 0; e < 8; e++) z[e] = 0.f;
      z = wmma_bf16(qf0.b, k0.b, z);
      s[t] = wmma_bf16(qf1.b, k1.b, z);
    }
    // ---- issue V fragment loads early: they overlap the softmax VALU ----
    FragU vf0, vf1, vf2, vf3;
    {
      const unsigned short* vr0 = vb + (size_t)(l16) * 2048 + j0 + 8 * hl;
      const unsigned short* vr1 = vr0 + (size_t)16 * 2048;
      const unsigned short* vr2 = vr1 + (size_t)16 * 2048;
      const unsigned short* vr3 = vr2 + (size_t)16 * 2048;
      vf0.u[0] = *(const v4u*)(vr0); vf0.u[1] = *(const v4u*)(vr0 + 16);
      vf1.u[0] = *(const v4u*)(vr1); vf1.u[1] = *(const v4u*)(vr1 + 16);
      vf2.u[0] = *(const v4u*)(vr2); vf2.u[1] = *(const v4u*)(vr2 + 16);
      vf3.u[0] = *(const v4u*)(vr3); vf3.u[1] = *(const v4u*)(vr3 + 16);
    }
    // ---- packed mask + scale: one dword covers 32 cols of one row ----
    unsigned int mbits[8];
#pragma unroll
    for (int v = 0; v < 8; v++)
      mbits[v] = pmrow[(size_t)(8 * hl + v) * 64 + (j0 >> 5)];
#pragma unroll
    for (int t = 0; t < 2; t++)
#pragma unroll
      for (int v = 0; v < 8; v++) {
        bool mk = (mbits[v] >> (16 * t + l16)) & 1u;
        float sv = s[t][v];
        s[t][v] = mk ? sv * scale : -1e30f;
      }
    // ---- online softmax (rows live in 16-lane halves of each C VGPR) ----
#pragma unroll
    for (int v = 0; v < 8; v++) {
      float mx = fmaxf(s[0][v], s[1][v]);
#pragma unroll
      for (int o_ = 1; o_ < 16; o_ <<= 1) mx = fmaxf(mx, __shfl_xor(mx, o_, 32));
      float mnew = fmaxf(mrow[v], mx);
      float corr = __expf(mrow[v] - mnew);
      float p0 = __expf(s[0][v] - mnew);
      float p1 = __expf(s[1][v] - mnew);
      s[0][v] = p0;
      s[1][v] = p1;
      float rs = p0 + p1;
#pragma unroll
      for (int o_ = 1; o_ < 16; o_ <<= 1) rs += __shfl_xor(rs, o_, 32);
      lrow[v] = lrow[v] * corr + rs;
      mrow[v] = mnew;
      acc0[v] *= corr; acc1[v] *= corr; acc2[v] *= corr; acc3[v] *= corr;
    }
    // ---- P (C layout) -> LDS -> A-frag layout ----
#pragma unroll
    for (int t = 0; t < 2; t++)
#pragma unroll
      for (int v = 0; v < 8; v++)
        plds[wv][(8 * hl + v) * 32 + 16 * t + l16] = f2bf(s[t][v]);
    __syncthreads();
    FragU pa;
    const unsigned short* prow = &plds[wv][l16 * 32 + 8 * hl];
    pa.u[0] = *(const v4u*)(prow);
    pa.u[1] = *(const v4u*)(prow + 16);
    // ---- O += P @ V ----
    acc0 = wmma_bf16(pa.b, vf0.b, acc0);
    acc1 = wmma_bf16(pa.b, vf1.b, acc1);
    acc2 = wmma_bf16(pa.b, vf2.b, acc2);
    acc3 = wmma_bf16(pa.b, vf3.b, acc3);
  }
  // ---- epilogue: 1/l, sigmoid gate, store bf16 O (B*NA, 512) ----
  const unsigned short* sgp = sg + ((size_t)(b * 8 + h) * 2048 + i0) * 64;
  unsigned short* op = o + ((size_t)(b * 2048) + i0) * 512 + h * 64;
  auto emit = [&](v8f accv, int t) {
#pragma unroll
    for (int v = 0; v < 8; v++) {
      float rl = 1.0f / lrow[v];
      int row = 8 * hl + v;
      int d = 16 * t + l16;
      float gate = bf2f(sgp[(size_t)row * 64 + d]);
      op[(size_t)row * 512 + d] = f2bf(accv[v] * rl * gate);
    }
  };
  emit(acc0, 0);
  emit(acc1, 1);
  emit(acc2, 2);
  emit(acc3, 3);
}

// ---------------------------------------------------------------------------
extern "C" void kernel_launch(void* const* d_in, const int* in_sizes, int n_in,
                              void* d_out, int out_size, void* d_ws,
                              size_t ws_size, hipStream_t stream) {
  const float* feat_a = (const float*)d_in[0];
  const float* feat_b = (const float*)d_in[1];
  const unsigned char* mask = (const unsigned char*)d_in[2];
  const float* Wq = (const float*)d_in[3];
  const float* bq = (const float*)d_in[4];
  const float* Wk = (const float*)d_in[5];
  const float* bk = (const float*)d_in[6];
  const float* Wv = (const float*)d_in[7];
  const float* bv = (const float*)d_in[8];
  const float* Wg = (const float*)d_in[9];
  const float* bg = (const float*)d_in[10];
  const float* Wo = (const float*)d_in[11];
  const float* bo = (const float*)d_in[12];
  const float* ln_a_w = (const float*)d_in[13];
  const float* ln_a_b = (const float*)d_in[14];
  const float* ln_b_w = (const float*)d_in[15];
  const float* ln_b_b = (const float*)d_in[16];
  const float* qln_w = (const float*)d_in[17];
  const float* qln_b = (const float*)d_in[18];
  const float* kln_w = (const float*)d_in[19];
  const float* kln_b = (const float*)d_in[20];

  char* ws = (char*)d_ws;
  size_t off = 0;
  auto alloc = [&](size_t bytes) {
    size_t o = off;
    off = (off + bytes + 255) & ~(size_t)255;
    return o;
  };
  unsigned short* fa = (unsigned short*)(ws + alloc(4096u * 256 * 2));
  unsigned short* fb = (unsigned short*)(ws + alloc(4096u * 256 * 2));
  unsigned short* wqT = (unsigned short*)(ws + alloc(512u * 256 * 2));
  unsigned short* wkT = (unsigned short*)(ws + alloc(512u * 256 * 2));
  unsigned short* wvT = (unsigned short*)(ws + alloc(512u * 256 * 2));
  unsigned short* wgT = (unsigned short*)(ws + alloc(512u * 256 * 2));
  unsigned short* woT = (unsigned short*)(ws + alloc(256u * 512 * 2));
  unsigned int* pmask = (unsigned int*)(ws + alloc(262144u * 4));
  float* qpre = (float*)(ws + alloc(4096u * 512 * 4));
  float* kpre = (float*)(ws + alloc(4096u * 512 * 4));
  unsigned short* qb = (unsigned short*)(ws + alloc(16u * 2048 * 64 * 2));
  unsigned short* kbuf = (unsigned short*)(ws + alloc(16u * 2048 * 64 * 2));
  unsigned short* vtb = (unsigned short*)(ws + alloc(16u * 64 * 2048 * 2));
  unsigned short* sgb = (unsigned short*)(ws + alloc(16u * 2048 * 64 * 2));
  unsigned short* ob = (unsigned short*)qpre;  // reuse: qpre dead after LN

  // 1) weights -> bf16 transposed; mask -> bits
  k_wT<<<512, 256, 0, stream>>>(Wq, wqT, 256, 512);
  k_wT<<<512, 256, 0, stream>>>(Wk, wkT, 256, 512);
  k_wT<<<512, 256, 0, stream>>>(Wv, wvT, 256, 512);
  k_wT<<<512, 256, 0, stream>>>(Wg, wgT, 256, 512);
  k_wT<<<512, 256, 0, stream>>>(Wo, woT, 512, 256);
  k_packmask<<<1024, 256, 0, stream>>>(mask, pmask, 262144);
  // 2) input LayerNorms -> bf16
  k_ln_feat<<<1024, 128, 0, stream>>>(feat_a, ln_a_w, ln_a_b, fa, 4096);
  k_ln_feat<<<1024, 128, 0, stream>>>(feat_b, ln_b_w, ln_b_b, fb, 4096);
  // 3) projections (4096x512x256 each), 32x32 tile per wave
  k_gemm32<<<512, 128, 0, stream>>>(fa, wqT, bq, qpre, 4096, 512, 256, 0);
  k_gemm32<<<512, 128, 0, stream>>>(fb, wkT, bk, kpre, 4096, 512, 256, 0);
  k_gemm32<<<512, 128, 0, stream>>>(fb, wvT, bv, vtb, 4096, 512, 256, 1);
  k_gemm32<<<512, 128, 0, stream>>>(fa, wgT, bg, sgb, 4096, 512, 256, 2);
  // 4) q/k LayerNorm + scatter to (B,H,N,64)
  k_ln_qk<<<1024, 128, 0, stream>>>(qpre, qln_w, qln_b, qb, 4096);
  k_ln_qk<<<1024, 128, 0, stream>>>(kpre, kln_w, kln_b, kbuf, 4096);
  // 5) flash attention + gate -> O bf16
  k_flash<<<512, 128, 0, stream>>>(qb, kbuf, vtb, sgb, pmask, ob);
  // 6) output projection -> f32 d_out (4096x256x512)
  k_gemm32<<<256, 128, 0, stream>>>(ob, woT, bo, d_out, 4096, 256, 512, 0);
}
